// PriorSoftmaxAttention_15616501088532
// MI455X (gfx1250) — matrix operand order repaired and checked by reading, not compiled, not run on hardware
//
#include <hip/hip_runtime.h>

typedef _Float16 v16h __attribute__((ext_vector_type(16)));
typedef float    v8f  __attribute__((ext_vector_type(8)));

#define WMMA_F32_F16(a, b, c) \
  __builtin_amdgcn_wmma_f32_16x16x32_f16(false, (a), false, (b), (short)0, (c), false, false)

namespace {
constexpr int   S       = 2048;
constexpr int   D       = 64;
constexpr int   H       = 16;
constexpr int   SP      = 2052;        // padded LDS row stride (floats); 16B-aligned rows
constexpr float SCALE   = 0.125f;      // 1/sqrt(64)
constexpr float LAMBDA  = 0.01f;
constexpr float MASKVAL = -1.0e7f;     // LAMBDA * (-1e9)
constexpr int   PTOFF   = 16;          // table shift: index = delta + PTOFF
constexpr int   PTN     = S + PTOFF;   // 2064 entries, delta in [-16, 2047]
}

__global__ __launch_bounds__(128)
void prior_softmax_attn_kernel(const float* __restrict__ Q,
                               const float* __restrict__ K,
                               const float* __restrict__ V,
                               float* __restrict__ Out,
                               float* __restrict__ Attn) {
  extern __shared__ float smem[];
  float* scores = smem;                    // 16 * SP floats
  float* obuf   = smem + 16 * SP;          // 4 waves * 16 * 64 floats (O partials)
  float* red    = obuf + 4 * 1024;         // 128 floats (row reductions)
  float* rowinv = red + 128;               // 16 floats
  float* ptabx  = rowinv + 16;             // PTN floats: prior(delta) incl. mask region

  const int tid   = threadIdx.x;
  const int wave  = tid >> 5;
  const int lane  = tid & 31;
  const int g     = lane >> 4;             // lane group (0: lanes 0-15, 1: lanes 16-31)
  const int ln    = lane & 15;
  const int head  = blockIdx.y;
  const int qb    = blockIdx.x;            // q-tile index
  const int qbase = qb << 4;
  const int nkc   = (qb + 1) << 4;         // active (unmasked) key columns
  const int nkc32 = (nkc + 31) & ~31;      // rounded up to WMMA K-chunk

  const float* Qh = Q + (size_t)head * S * D;
  const float* Kh = K + (size_t)head * S * D;
  const float* Vh = V + (size_t)head * S * D;

  // ---- Phase 0: shifted prior table; negative deltas hold the causal mask ----
  for (int i = tid; i < PTN; i += 128) {
    const int dlt = i - PTOFF;
    ptabx[i] = (dlt >= 0) ? (-LAMBDA * __logf((float)dlt + 1.0f)) : MASKVAL;
  }

  // ---- Load Q A-fragments (16x32 f16, K-dim = d) once per wave ----
  v16h aq0, aq1;
  {
    const float* qr = Qh + (size_t)(qbase + ln) * D;   // A: M = lane&15
    #pragma unroll
    for (int h = 0; h < 16; ++h) {
      const int d = (g << 4) + h;   // A-frag K mapping (contiguous: k = 16g + h)
      aq0[h] = (_Float16)qr[d];
      aq1[h] = (_Float16)qr[d + 32];
    }
  }
  __syncthreads();   // ptabx visible before phase 1 reads it

  // ---- Phase 1: causal tiles only: scores = scale*(Q K^T) + lambda*prior ----
  for (int kt = wave; kt <= qb; kt += 4) {
    const int kbase = kt << 4;
    const float* kr = Kh + (size_t)(kbase + ln) * D + (g << 4);
    v16h b0, b1;
    #pragma unroll
    for (int h = 0; h < 16; ++h) {
      b0[h] = (_Float16)kr[h];
      b1[h] = (_Float16)kr[h + 32];
    }
    v8f c = {};
    c = WMMA_F32_F16(aq0, b0, c);
    c = WMMA_F32_F16(aq1, b1, c);

    // prior lookup: 8 consecutive LDS addresses, no compares, no branches
    const float* pbase = ptabx + (qbase + (g << 3) + PTOFF - (kbase + ln));
    float pt[8];
    #pragma unroll
    for (int r = 0; r < 8; ++r) pt[r] = pbase[r];
    #pragma unroll
    for (int r = 0; r < 8; ++r)
      scores[(r + (g << 3)) * SP + kbase + ln] = c[r] * SCALE + pt[r];
  }
  __syncthreads();

  // ---- Phase 2: row softmax stats over active columns (8 threads/row) ----
  {
    const int r  = tid >> 3;
    const int s8 = tid & 7;
    float* srow  = scores + r * SP;

    float mx = -3.0e38f;
    for (int c = s8 << 2; c < nkc; c += 32) {
      const float4 v = *(const float4*)&srow[c];
      mx = fmaxf(mx, fmaxf(fmaxf(v.x, v.y), fmaxf(v.z, v.w)));
    }
    red[tid] = mx;
    __syncthreads();
    float rmx = red[r << 3];
    #pragma unroll
    for (int j = 1; j < 8; ++j) rmx = fmaxf(rmx, red[(r << 3) + j]);
    __syncthreads();

    float sum = 0.0f;
    for (int c = s8 << 2; c < nkc; c += 32) {
      float4 v = *(float4*)&srow[c];
      v.x = __expf(v.x - rmx);
      v.y = __expf(v.y - rmx);
      v.z = __expf(v.z - rmx);
      v.w = __expf(v.w - rmx);
      *(float4*)&srow[c] = v;
      sum += (v.x + v.y) + (v.z + v.w);
    }
    red[tid] = sum;
    __syncthreads();
    float rs = 0.0f;
    #pragma unroll
    for (int j = 0; j < 8; ++j) rs += red[(r << 3) + j];
    if (s8 == 0) rowinv[r] = 1.0f / rs;
  }
  __syncthreads();

  // ---- Phase 3: normalize + store attn (active), zeros for masked region ----
  {
    float* ab = Attn + ((size_t)head * S + qbase) * S;
    const float4 z = make_float4(0.0f, 0.0f, 0.0f, 0.0f);
    for (int rr = 0; rr < 16; ++rr) {
      const float inv = rowinv[rr];
      float* srow = scores + rr * SP;
      float* arow = ab + (size_t)rr * S;
      for (int c4 = tid << 2; c4 < nkc; c4 += 512) {
        float4 v = *(float4*)&srow[c4];
        v.x *= inv; v.y *= inv; v.z *= inv; v.w *= inv;
        *(float4*)&srow[c4] = v;
        *(float4*)&arow[c4] = v;
      }
      for (int c4 = nkc + (tid << 2); c4 < nkc32; c4 += 512)   // pad P for last chunk
        *(float4*)&srow[c4] = z;
      for (int c4 = nkc + (tid << 2); c4 < S; c4 += 512)       // masked attn = 0
        *(float4*)&arow[c4] = z;
    }
  }
  __syncthreads();

  // ---- Phase 4: O = P @ V via WMMA over active k-chunks, per-wave partials ----
  {
    const int nj = nkc32 >> 5;            // active 32-wide K chunks
    v8f o0 = {}, o1 = {}, o2 = {}, o3 = {};
    for (int j = wave; j < nj; j += 4) {
      const int kb = j << 5;
      v16h ap;
      #pragma unroll
      for (int h = 0; h < 16; ++h) {
        const int kk = kb + (g << 4) + h;
        ap[h] = (_Float16)scores[ln * SP + kk];      // A: M = lane&15
      }
      const float* vb = Vh + (size_t)(kb + (g << 4)) * D + ln;
      v16h bv;
      #pragma unroll
      for (int h = 0; h < 16; ++h) bv[h] = (_Float16)vb[h * D];
      o0 = WMMA_F32_F16(ap, bv, o0);
      #pragma unroll
      for (int h = 0; h < 16; ++h) bv[h] = (_Float16)vb[h * D + 16];
      o1 = WMMA_F32_F16(ap, bv, o1);
      #pragma unroll
      for (int h = 0; h < 16; ++h) bv[h] = (_Float16)vb[h * D + 32];
      o2 = WMMA_F32_F16(ap, bv, o2);
      #pragma unroll
      for (int h = 0; h < 16; ++h) bv[h] = (_Float16)vb[h * D + 48];
      o3 = WMMA_F32_F16(ap, bv, o3);
    }
    float* ob = obuf + wave * 1024;
    #pragma unroll
    for (int r = 0; r < 8; ++r) {
      const int m = r + (g << 3);
      ob[m * 64 + ln]      = o0[r];
      ob[m * 64 + 16 + ln] = o1[r];
      ob[m * 64 + 32 + ln] = o2[r];
      ob[m * 64 + 48 + ln] = o3[r];
    }
  }
  __syncthreads();

  // ---- Deterministic cross-wave sum + vectorized store of O ----
  {
    float* outp = Out + ((size_t)head * S + qbase) * D;
    for (int idx = tid; idx < 256; idx += 128) {
      const int rr = idx >> 4;
      const int dd = (idx & 15) << 2;
      const int o  = rr * 64 + dd;
      const float4 a = *(const float4*)&obuf[o];
      const float4 b = *(const float4*)&obuf[1024 + o];
      const float4 c = *(const float4*)&obuf[2048 + o];
      const float4 d = *(const float4*)&obuf[3072 + o];
      float4 s;
      s.x = (a.x + b.x) + (c.x + d.x);
      s.y = (a.y + b.y) + (c.y + d.y);
      s.z = (a.z + b.z) + (c.z + d.z);
      s.w = (a.w + b.w) + (c.w + d.w);
      *(float4*)&outp[(size_t)rr * D + dd] = s;
    }
  }
}

extern "C" void kernel_launch(void* const* d_in, const int* in_sizes, int n_in,
                              void* d_out, int out_size, void* d_ws, size_t ws_size,
                              hipStream_t stream) {
  (void)in_sizes; (void)n_in; (void)out_size; (void)d_ws; (void)ws_size;
  const float* Q = (const float*)d_in[0];
  const float* K = (const float*)d_in[1];
  const float* V = (const float*)d_in[2];
  float* Out  = (float*)d_out;                       // [1,16,2048,64] first
  float* Attn = Out + (size_t)H * S * D;             // then [1,16,2048,2048]

  const size_t shmem = (size_t)(16 * SP + 4 * 1024 + 128 + 16 + PTN) * sizeof(float);
  dim3 grid(S / 16, H, 1);   // 128 q-tiles x 16 heads
  dim3 block(128, 1, 1);     // 4 wave32
  prior_softmax_attn_kernel<<<grid, block, shmem, stream>>>(Q, K, V, Out, Attn);
}
